// Decoder_51599737094426
// MI455X (gfx1250) — matrix operand order repaired and checked by reading, not compiled, not run on hardware
//
#include <hip/hip_runtime.h>
#include <hip/hip_bf16.h>

// ---------------- constants (match reference shapes) ----------------
#define BB   32
#define SS   128
#define HH   1024
#define EE   256
#define VV   16000
#define TT   48
#define VS   16128          // V + S
#define TOT  32128          // 2V + S
#define GRUIN 2304          // H + H + E
#define NEGV (-1000000.0f)

typedef unsigned short u16;
typedef unsigned int   u32;

typedef __attribute__((ext_vector_type(16))) __bf16 v16bf;
typedef __attribute__((ext_vector_type(8)))  float  v8f;
typedef __attribute__((ext_vector_type(4)))  float  f32x4;

union Pack32 { f32x4 q[2]; v16bf v; };

__device__ __forceinline__ u16 f2bf(float f) {
    u32 u = __float_as_uint(f);
    u += 0x7FFFu + ((u >> 16) & 1u);   // round-to-nearest-even
    return (u16)(u >> 16);
}

// ---------------- fp32 -> bf16 weight conversion ----------------
__global__ __launch_bounds__(256) void cvt_bf16(const float* __restrict__ src,
                                                u16* __restrict__ dst, int n) {
    int i = blockIdx.x * 256 + threadIdx.x;
    if (i < n) dst[i] = f2bf(src[i]);
}

// ---------------- generic skinny GEMM: C[32,N] = A_bf[32,K] * B_bf[N,K]^T + bias ----
// 256 threads = 8 waves; wave w owns one 16-wide n-tile (n0 = blk*128 + w*16) and
// computes BOTH m-tiles (32x16 output), so the B (weight) block is fetched once per
// two WMMAs. K-loop is software-pipelined (double-buffered) so global_load_b128 of
// block i+1 overlaps v_wmma on block i (partial s_wait_loadcnt instead of 0).
// Per-lane addressing matches CDNA5 ISA 7.12.2 16-bit A(16x32) / B(32x16) layouts.
__global__ __launch_bounds__(256) void gemm_bf16(const u16* __restrict__ A,
                                                 const u16* __restrict__ B,
                                                 const float* __restrict__ bias,
                                                 float* __restrict__ C,
                                                 int N, int K) {
    const int lane = threadIdx.x & 31;
    const int wave = threadIdx.x >> 5;
    const int n0   = (blockIdx.x * 8 + wave) * 16;
    const int half = lane >> 4;                   // lane 0-15 / 16-31
    const int l    = lane & 15;

    // A 16x32 bf16 tile: v0-3 = K[half*8 .. +7], v4-7 = K[16+half*8 .. +7]
    const u16* a0p = A + (size_t)l * K + half * 8;          // m-tile 0 (rows 0..15)
    const u16* a1p = A + (size_t)(16 + l) * K + half * 8;   // m-tile 1 (rows 16..31)
    // B 32x16 bf16 tile: n = l, v0-3 = K[half*8 .. +7], v4-7 = K[16+half*8 .. +7]
    const u16* bp  = B + (size_t)(n0 + l) * K + half * 8;

    v8f c0 = {}, c1 = {};
    Pack32 A0s0, A1s0, Bs0, A0s1, A1s1, Bs1;

    // preload K-block 0
    A0s0.q[0] = *(const f32x4*)(a0p);      A0s0.q[1] = *(const f32x4*)(a0p + 16);
    A1s0.q[0] = *(const f32x4*)(a1p);      A1s0.q[1] = *(const f32x4*)(a1p + 16);
    Bs0.q[0]  = *(const f32x4*)(bp);       Bs0.q[1]  = *(const f32x4*)(bp + 16);

    const int nb = K >> 5;                 // 32-wide K blocks (nb even for all K here)
    int i = 1;
    for (; i + 1 < nb; i += 2) {
        const int k1 = i << 5, k2 = (i + 1) << 5;
        if (k1 + 128 < K) __builtin_prefetch(bp + k1 + 128, 0, 3);  // global_prefetch_b8
        // stage 1 <- block i, while computing block i-1
        A0s1.q[0] = *(const f32x4*)(a0p + k1); A0s1.q[1] = *(const f32x4*)(a0p + k1 + 16);
        A1s1.q[0] = *(const f32x4*)(a1p + k1); A1s1.q[1] = *(const f32x4*)(a1p + k1 + 16);
        Bs1.q[0]  = *(const f32x4*)(bp + k1);  Bs1.q[1]  = *(const f32x4*)(bp + k1 + 16);
        c0 = __builtin_amdgcn_wmma_f32_16x16x32_bf16(false, A0s0.v, false, Bs0.v, (short)0, c0, false, false);
        c1 = __builtin_amdgcn_wmma_f32_16x16x32_bf16(false, A1s0.v, false, Bs0.v, (short)0, c1, false, false);
        // stage 0 <- block i+1, while computing block i
        A0s0.q[0] = *(const f32x4*)(a0p + k2); A0s0.q[1] = *(const f32x4*)(a0p + k2 + 16);
        A1s0.q[0] = *(const f32x4*)(a1p + k2); A1s0.q[1] = *(const f32x4*)(a1p + k2 + 16);
        Bs0.q[0]  = *(const f32x4*)(bp + k2);  Bs0.q[1]  = *(const f32x4*)(bp + k2 + 16);
        c0 = __builtin_amdgcn_wmma_f32_16x16x32_bf16(false, A0s1.v, false, Bs1.v, (short)0, c0, false, false);
        c1 = __builtin_amdgcn_wmma_f32_16x16x32_bf16(false, A1s1.v, false, Bs1.v, (short)0, c1, false, false);
    }
    if (i < nb) {                          // one trailing block (i == nb-1)
        const int k1 = i << 5;
        A0s1.q[0] = *(const f32x4*)(a0p + k1); A0s1.q[1] = *(const f32x4*)(a0p + k1 + 16);
        A1s1.q[0] = *(const f32x4*)(a1p + k1); A1s1.q[1] = *(const f32x4*)(a1p + k1 + 16);
        Bs1.q[0]  = *(const f32x4*)(bp + k1);  Bs1.q[1]  = *(const f32x4*)(bp + k1 + 16);
        c0 = __builtin_amdgcn_wmma_f32_16x16x32_bf16(false, A0s0.v, false, Bs0.v, (short)0, c0, false, false);
        c1 = __builtin_amdgcn_wmma_f32_16x16x32_bf16(false, A1s0.v, false, Bs0.v, (short)0, c1, false, false);
        c0 = __builtin_amdgcn_wmma_f32_16x16x32_bf16(false, A0s1.v, false, Bs1.v, (short)0, c0, false, false);
        c1 = __builtin_amdgcn_wmma_f32_16x16x32_bf16(false, A1s1.v, false, Bs1.v, (short)0, c1, false, false);
    } else {
        c0 = __builtin_amdgcn_wmma_f32_16x16x32_bf16(false, A0s0.v, false, Bs0.v, (short)0, c0, false, false);
        c1 = __builtin_amdgcn_wmma_f32_16x16x32_bf16(false, A1s0.v, false, Bs0.v, (short)0, c1, false, false);
    }

    const float bv = bias ? bias[n0 + l] : 0.0f;
    // D layout: row m = (mtile*16) + half*8 + v, col = n0 + l
    float* cp0 = C + (size_t)(half * 8) * N + n0 + l;
    float* cp1 = C + (size_t)(16 + half * 8) * N + n0 + l;
#pragma unroll
    for (int v = 0; v < 8; ++v) { cp0[(size_t)v * N] = c0[v] + bv; cp1[(size_t)v * N] = c1[v] + bv; }
}

// ---------------- attention scores + softmax + context (one block per batch) ----------------
__global__ __launch_bounds__(256) void attn_context(const float* __restrict__ th,
                                                    const float* __restrict__ enc,
                                                    u16* __restrict__ x_bf) {
    const int b = blockIdx.x, tid = threadIdx.x;
    __shared__ float sth[HH];
    __shared__ float sc[SS];
    __shared__ float red[256];
    for (int i = tid; i < HH; i += 256) sth[i] = th[b * HH + i];
    __syncthreads();
    const int wave = tid >> 5, lane = tid & 31;
    for (int s = wave; s < SS; s += 8) {
        const float* e = enc + ((size_t)b * SS + s) * HH;
        float acc = 0.f;
        for (int k = lane; k < HH; k += 32) acc += e[k] * sth[k];
        for (int off = 16; off; off >>= 1) acc += __shfl_down(acc, off, 32);
        if (!lane) sc[s] = acc;
    }
    __syncthreads();
    float v = (tid < SS) ? sc[tid] : -3.4e38f;
    red[tid] = v; __syncthreads();
    for (int off = 128; off; off >>= 1) { if (tid < off) red[tid] = fmaxf(red[tid], red[tid + off]); __syncthreads(); }
    const float mx = red[0]; __syncthreads();
    float e0 = (tid < SS) ? __expf(sc[tid] - mx) : 0.f;
    red[tid] = e0; __syncthreads();
    for (int off = 128; off; off >>= 1) { if (tid < off) red[tid] += red[tid + off]; __syncthreads(); }
    const float Z = red[0]; __syncthreads();
    if (tid < SS) sc[tid] = e0 / Z;
    __syncthreads();
    float a0 = 0.f, a1 = 0.f, a2 = 0.f, a3 = 0.f;
    for (int s = 0; s < SS; ++s) {
        const float a = sc[s];
        const float* e = enc + ((size_t)b * SS + s) * HH;
        a0 += a * e[tid]; a1 += a * e[tid + 256]; a2 += a * e[tid + 512]; a3 += a * e[tid + 768];
    }
    u16* xr = x_bf + b * GRUIN;
    xr[tid] = f2bf(a0); xr[tid + 256] = f2bf(a1); xr[tid + 512] = f2bf(a2); xr[tid + 768] = f2bf(a3);
}

// ---------------- copy scores: cseq[b,s] = dot(enc[b,s], th2[b]) ----------------
__global__ __launch_bounds__(256) void copy_scores_kernel(const float* __restrict__ th2,
                                                          const float* __restrict__ enc,
                                                          float* __restrict__ cseq) {
    const int b = blockIdx.x, tid = threadIdx.x;
    __shared__ float sth[HH];
    for (int i = tid; i < HH; i += 256) sth[i] = th2[b * HH + i];
    __syncthreads();
    const int wave = tid >> 5, lane = tid & 31;
    for (int s = wave; s < SS; s += 8) {
        const float* e = enc + ((size_t)b * SS + s) * HH;
        float acc = 0.f;
        for (int k = lane; k < HH; k += 32) acc += e[k] * sth[k];
        for (int off = 16; off; off >>= 1) acc += __shfl_down(acc, off, 32);
        if (!lane) cseq[b * SS + s] = acc;
    }
}

// ---------------- GRU pointwise update ----------------
__global__ __launch_bounds__(256) void gru_update(const float* __restrict__ gx,
                                                  const float* __restrict__ gh,
                                                  float* __restrict__ h,
                                                  u16* __restrict__ h_bf) {
    const int i = blockIdx.x * 256 + threadIdx.x;   // 32*1024 total
    const int b = i >> 10, j = i & 1023;
    const float* gxr = gx + (size_t)b * 3 * HH;
    const float* ghr = gh + (size_t)b * 3 * HH;
    const float r = 1.f / (1.f + __expf(-(gxr[j] + ghr[j])));
    const float z = 1.f / (1.f + __expf(-(gxr[HH + j] + ghr[HH + j])));
    const float n = tanhf(gxr[2 * HH + j] + r * ghr[2 * HH + j]);
    const float hn = (1.f - z) * n + z * h[i];
    h[i] = hn;
    h_bf[i] = f2bf(hn);
}

// ---------------- copy-score scatter into extended vocab ----------------
__global__ __launch_bounds__(256) void zero_cs(float* __restrict__ cs, int* __restrict__ cnt) {
    const int i = blockIdx.x * 256 + threadIdx.x;
    if (i < BB * VS) { cs[i] = 0.f; cnt[i] = 0; }
}
__global__ __launch_bounds__(128) void scatter_copy(const int* __restrict__ inp,
                                                    const float* __restrict__ cseq,
                                                    float* __restrict__ cs, int* __restrict__ cnt) {
    const int b = blockIdx.x, s = threadIdx.x;
    const int tok = inp[b * SS + s];
    atomicAdd(cs + (size_t)b * VS + tok, cseq[b * SS + s]);
    atomicAdd(cnt + (size_t)b * VS + tok, 1);
}

// ---------------- per-step output: mixture softmax, log-probs, argmax, selective read ----------------
__global__ __launch_bounds__(256) void output_step(const float* __restrict__ gen,
                                                   const float* __restrict__ cs,
                                                   const int* __restrict__ cnt,
                                                   const float* __restrict__ cseq,
                                                   const int* __restrict__ inp,
                                                   const float* __restrict__ enc,
                                                   const u16* __restrict__ emb_bf,
                                                   float* __restrict__ out,
                                                   u16* __restrict__ x_bf,
                                                   int* __restrict__ smp, int t) {
    const int b = blockIdx.x, tid = threadIdx.x;
    const float* genr = gen + (size_t)b * VV;
    const float* csr  = cs + (size_t)b * VS;
    const int*   cntr = cnt + (size_t)b * VS;
    __shared__ float red[256];
    __shared__ int   redi[256];
    __shared__ float snorm[SS];
    __shared__ float s_mx, s_Z, s_den;
    __shared__ int   s_arg;

    auto logit = [&](int j) -> float {
        if (j < VV) return (j == 0) ? NEGV : genr[j];
        const int jj = j - VV;
        return (jj == 0 || cntr[jj] == 0) ? NEGV : csr[jj];
    };
    // pass 1: max over concat(gen, copy_scores)
    float mx = -3.4e38f;
    for (int j = tid; j < TOT; j += 256) mx = fmaxf(mx, logit(j));
    red[tid] = mx; __syncthreads();
    for (int off = 128; off; off >>= 1) { if (tid < off) red[tid] = fmaxf(red[tid], red[tid + off]); __syncthreads(); }
    if (!tid) s_mx = red[0]; __syncthreads();
    mx = s_mx;
    // pass 2: partition function
    float sum = 0.f;
    for (int j = tid; j < TOT; j += 256) sum += __expf(logit(j) - mx);
    red[tid] = sum; __syncthreads();
    for (int off = 128; off; off >>= 1) { if (tid < off) red[tid] += red[tid + off]; __syncthreads(); }
    if (!tid) s_Z = red[0]; __syncthreads();
    const float invZ = 1.f / s_Z;
    // pass 3: final probs, log-probs, argmax
    float bestp = -1.f; int bestj = 0x7fffffff;
    float* outr = out + ((size_t)t * BB + b) * VS;
    for (int jj = tid; jj < VS; jj += 256) {
        const float pg = (jj < VV && jj != 0) ? __expf(genr[jj] - mx) : 0.f;
        const float pl = (jj == 0 || cntr[jj] == 0) ? NEGV : csr[jj];
        const float p  = (pg + __expf(pl - mx)) * invZ;
        outr[jj] = __logf(p + 1e-10f);
        if (p > bestp || (p == bestp && jj < bestj)) { bestp = p; bestj = jj; }
    }
    red[tid] = bestp; redi[tid] = bestj; __syncthreads();
    for (int off = 128; off; off >>= 1) {
        if (tid < off) {
            const float po = red[tid + off]; const int jo = redi[tid + off];
            if (po > red[tid] || (po == red[tid] && jo < redi[tid])) { red[tid] = po; redi[tid] = jo; }
        }
        __syncthreads();
    }
    if (!tid) {
        s_arg = redi[0];
        smp[b] = redi[0];
        out[(size_t)TT * BB * VS + (size_t)b * TT + t] = (float)redi[0];
    }
    __syncthreads();
    const int samp = s_arg;
    // selective read for next step
    float selv = 0.f;
    if (tid < SS) selv = (inp[b * SS + tid] == samp) ? cseq[b * SS + tid] : 0.f;
    red[tid] = (tid < SS) ? fabsf(selv) : 0.f; __syncthreads();
    for (int off = 128; off; off >>= 1) { if (tid < off) red[tid] += red[tid + off]; __syncthreads(); }
    if (!tid) s_den = fmaxf(red[0], 1e-12f); __syncthreads();
    if (tid < SS) snorm[tid] = selv / s_den;
    __syncthreads();
    float a0 = 0.f, a1 = 0.f, a2 = 0.f, a3 = 0.f;
    for (int s = 0; s < SS; ++s) {
        const float a = snorm[s];
        const float* e = enc + ((size_t)b * SS + s) * HH;
        a0 += a * e[tid]; a1 += a * e[tid + 256]; a2 += a * e[tid + 512]; a3 += a * e[tid + 768];
    }
    u16* xr = x_bf + b * GRUIN;
    xr[HH + tid] = f2bf(a0); xr[HH + tid + 256] = f2bf(a1);
    xr[HH + tid + 512] = f2bf(a2); xr[HH + tid + 768] = f2bf(a3);
    // embedding of sampled token for next step (OOV -> UNK, clamp like gather)
    int prev = (samp > VV) ? 3 : samp;
    if (prev >= VV) prev = VV - 1;
    xr[2 * HH + tid] = emb_bf[(size_t)prev * EE + tid];   // blockDim == EE == 256
}

// ---------------- init kernels ----------------
__global__ __launch_bounds__(256) void init_state(const float* __restrict__ hidden,
                                                  float* __restrict__ h, u16* __restrict__ h_bf) {
    const int i = blockIdx.x * 256 + threadIdx.x;     // 32*1024
    const float v = hidden[i];
    h[i] = v; h_bf[i] = f2bf(v);
}
__global__ __launch_bounds__(256) void sos_init(float* __restrict__ out) {
    const int i = blockIdx.x * 256 + threadIdx.x;
    if (i < BB * VS) out[i] = 2.0f;                   // SOS row (t=0)
}
__global__ __launch_bounds__(256) void init_misc(const u16* __restrict__ emb_bf,
                                                 u16* __restrict__ x_bf,
                                                 int* __restrict__ smp,
                                                 float* __restrict__ out) {
    const int i = blockIdx.x * 256 + threadIdx.x;
    if (i < BB * (HH + EE)) {
        const int b = i / (HH + EE), off = i % (HH + EE);
        if (off < HH) x_bf[b * GRUIN + HH + off] = 0;                           // sel_read = 0
        else          x_bf[b * GRUIN + 2 * HH + (off - HH)] = emb_bf[EE + (off - HH)]; // emb[1]
    }
    if (i < BB) {
        smp[i] = 1;
        out[(size_t)TT * BB * VS + (size_t)i * TT] = 1.0f;   // sampled_idxs[:,0] = 1
    }
}

// ---------------- host launcher ----------------
static inline size_t alignup(size_t x) { return (x + 255) & ~(size_t)255; }

extern "C" void kernel_launch(void* const* d_in, const int* in_sizes, int n_in,
                              void* d_out, int out_size, void* d_ws, size_t ws_size,
                              hipStream_t stream) {
    const int*   input   = (const int*)  d_in[0];
    const float* hidden  = (const float*)d_in[1];
    const float* enc     = (const float*)d_in[2];
    const float* attnW   = (const float*)d_in[4];
    const float* attnB   = (const float*)d_in[5];
    const float* copyW   = (const float*)d_in[6];
    const float* copyB   = (const float*)d_in[7];
    const float* embedF  = (const float*)d_in[8];
    const float* WihF    = (const float*)d_in[9];
    const float* WhhF    = (const float*)d_in[10];
    const float* bih     = (const float*)d_in[11];
    const float* bhh     = (const float*)d_in[12];
    const float* outWF   = (const float*)d_in[13];
    const float* outB    = (const float*)d_in[14];
    float* out = (float*)d_out;

    // workspace carve-up (all 256B aligned)
    char* w = (char*)d_ws; size_t o = 0;
    u16* attnW_bf = (u16*)(w + o); o = alignup(o + (size_t)HH * HH * 2);
    u16* copyW_bf = (u16*)(w + o); o = alignup(o + (size_t)HH * HH * 2);
    u16* wih_bf   = (u16*)(w + o); o = alignup(o + (size_t)3 * HH * GRUIN * 2);
    u16* whh_bf   = (u16*)(w + o); o = alignup(o + (size_t)3 * HH * HH * 2);
    u16* outw_bf  = (u16*)(w + o); o = alignup(o + (size_t)VV * HH * 2);
    u16* emb_bf   = (u16*)(w + o); o = alignup(o + (size_t)VV * EE * 2);
    float* h      = (float*)(w + o); o = alignup(o + (size_t)BB * HH * 4);
    u16* h_bf     = (u16*)(w + o); o = alignup(o + (size_t)BB * HH * 2);
    u16* x_bf     = (u16*)(w + o); o = alignup(o + (size_t)BB * GRUIN * 2);
    float* th     = (float*)(w + o); o = alignup(o + (size_t)BB * HH * 4);
    float* gx     = (float*)(w + o); o = alignup(o + (size_t)BB * 3 * HH * 4);
    float* gh     = (float*)(w + o); o = alignup(o + (size_t)BB * 3 * HH * 4);
    float* gen    = (float*)(w + o); o = alignup(o + (size_t)BB * VV * 4);
    float* cseq   = (float*)(w + o); o = alignup(o + (size_t)BB * SS * 4);
    float* cs     = (float*)(w + o); o = alignup(o + (size_t)BB * VS * 4);
    int*   cnt    = (int*)  (w + o); o = alignup(o + (size_t)BB * VS * 4);
    int*   smp    = (int*)  (w + o); o = alignup(o + (size_t)BB * 4);
    (void)ws_size; (void)in_sizes; (void)n_in; (void)out_size;

    // one-time per launch: bf16 weights, state, SOS row
    cvt_bf16<<<(HH * HH + 255) / 256, 256, 0, stream>>>(attnW, attnW_bf, HH * HH);
    cvt_bf16<<<(HH * HH + 255) / 256, 256, 0, stream>>>(copyW, copyW_bf, HH * HH);
    cvt_bf16<<<(3 * HH * GRUIN + 255) / 256, 256, 0, stream>>>(WihF, wih_bf, 3 * HH * GRUIN);
    cvt_bf16<<<(3 * HH * HH + 255) / 256, 256, 0, stream>>>(WhhF, whh_bf, 3 * HH * HH);
    cvt_bf16<<<(VV * HH + 255) / 256, 256, 0, stream>>>(outWF, outw_bf, VV * HH);
    cvt_bf16<<<(VV * EE + 255) / 256, 256, 0, stream>>>(embedF, emb_bf, VV * EE);
    init_state<<<(BB * HH) / 256, 256, 0, stream>>>(hidden, h, h_bf);
    sos_init<<<(BB * VS + 255) / 256, 256, 0, stream>>>(out);
    init_misc<<<(BB * (HH + EE) + 255) / 256, 256, 0, stream>>>(emb_bf, x_bf, smp, out);

    for (int t = 1; t < TT; ++t) {
        // attention query
        gemm_bf16<<<HH / 128, 256, 0, stream>>>(h_bf, attnW_bf, attnB, th, HH, HH);
        attn_context<<<BB, 256, 0, stream>>>(th, enc, x_bf);           // writes x[:, :1024]
        // GRU gates
        gemm_bf16<<<(3 * HH) / 128, 256, 0, stream>>>(x_bf, wih_bf, bih, gx, 3 * HH, GRUIN);
        gemm_bf16<<<(3 * HH) / 128, 256, 0, stream>>>(h_bf, whh_bf, bhh, gh, 3 * HH, HH);
        gru_update<<<(BB * HH) / 256, 256, 0, stream>>>(gx, gh, h, h_bf);
        // copy attention
        gemm_bf16<<<HH / 128, 256, 0, stream>>>(h_bf, copyW_bf, copyB, th, HH, HH);
        copy_scores_kernel<<<BB, 256, 0, stream>>>(th, enc, cseq);
        zero_cs<<<(BB * VS + 255) / 256, 256, 0, stream>>>(cs, cnt);
        scatter_copy<<<BB, SS, 0, stream>>>(input, cseq, cs, cnt);
        // generation logits (dominant GEMM: 32 x 16000 x 1024)
        gemm_bf16<<<VV / 128, 256, 0, stream>>>(h_bf, outw_bf, outB, gen, VV, HH);
        // mixture softmax + argmax + selective read + next-step x
        output_step<<<BB, 256, 0, stream>>>(gen, cs, cnt, cseq, input, enc, emb_bf,
                                            out, x_bf, smp, t);
    }
}
